// BRNN_30949534335195
// MI455X (gfx1250) — matrix operand order repaired and checked by reading, not compiled, not run on hardware
//
#include <hip/hip_runtime.h>
#include <hip/hip_bf16.h>

// ---------------- problem dims ----------------
#define HDIM   1024
#define HP1    1025
#define OUTD   32
#define ACTD   8
#define BSZ    128
#define TSTEPS 255
#define LOG2PI_F 1.8378770664093453f

typedef __attribute__((ext_vector_type(16))) _Float16 v16h;
typedef __attribute__((ext_vector_type(8)))  _Float16 v8h;
typedef __attribute__((ext_vector_type(8)))  float    v8f;

// ---------------- small helpers ----------------
__device__ inline float wave_reduce(float v) {
#pragma unroll
  for (int off = 16; off > 0; off >>= 1) v += __shfl_xor(v, off, 32);
  return v;
}

// =====================================================================
// Generic WMMA "NT" GEMM: C[M,N] = act( A[M,K] * W[N,K]^T + bias[N] )
// A, W fp16 row-major. One 16x16 tile per wave, K-loop step 32.
// Fragment packing per CDNA5 ISA 7.12.2 (16-bit A 16x32 / B 32x16 / f32 C).
// act: 0=none 1=relu 2=tanh. Optional fp16 mirror output Ch.
// =====================================================================
__global__ void gemm_nt_wmma(const _Float16* __restrict__ A, int lda,
                             const _Float16* __restrict__ W, int ldw,
                             const float* __restrict__ bias,
                             float* __restrict__ C, int ldc,
                             _Float16* __restrict__ Ch, int ldch,
                             int N, int K, int act) {
  const int wave  = threadIdx.x >> 5;
  const int lane  = threadIdx.x & 31;
  const int tileN = blockIdx.x * 4 + wave;   // wave-uniform
  const int tileM = blockIdx.y;
  if (tileN * 16 >= N) return;               // whole wave exits together

  const int m  = lane & 15;                  // A row within tile
  const int kg = (lane >> 4) << 3;           // A k-group base: 0 or 8
  const int kb = (lane >> 4) << 3;           // B k-group base: 0 or 8

  const _Float16* pa = A + (size_t)(tileM * 16 + m) * lda;
  const _Float16* pw = W + (size_t)(tileN * 16 + (lane & 15)) * ldw;

  v8f acc = {};
  for (int k0 = 0; k0 < K; k0 += 32) {
    v8h a0 = *(const v8h*)(pa + k0 + kg);
    v8h a1 = *(const v8h*)(pa + k0 + kg + 16);
    v8h b0 = *(const v8h*)(pw + k0 + kb);
    v8h b1 = *(const v8h*)(pw + k0 + kb + 16);
    v16h af = __builtin_shufflevector(a0, a1, 0,1,2,3,4,5,6,7,8,9,10,11,12,13,14,15);
    v16h bf = __builtin_shufflevector(b0, b1, 0,1,2,3,4,5,6,7,8,9,10,11,12,13,14,15);
    acc = __builtin_amdgcn_wmma_f32_16x16x32_f16(false, af, false, bf,
                                                 (short)0, acc, false, false);
  }

  const int n     = lane & 15;
  const int mbase = (lane >> 4) << 3;
  const int gn    = tileN * 16 + n;
  const float bv  = bias ? bias[gn] : 0.0f;
#pragma unroll
  for (int e = 0; e < 8; ++e) {
    int gm  = tileM * 16 + mbase + e;
    float v = acc[e] + bv;
    if (act == 1)      v = fmaxf(v, 0.0f);
    else if (act == 2) v = tanhf(v);
    C[(size_t)gm * ldc + gn] = v;
    if (Ch) Ch[(size_t)gm * ldch + gn] = (_Float16)v;
  }
}

// =====================================================================
// Prep: Wmain_h[b,i,j] = (half)(mu[i,j] + exp(0.5*lv[i,j]) * eps[b,i,j]), j<1024
// =====================================================================
__global__ void prep_w_kernel(const float* __restrict__ eps,
                              const float* __restrict__ mu,
                              const float* __restrict__ logvar,
                              _Float16* __restrict__ Wm) {
  size_t tid = (size_t)blockIdx.x * blockDim.x + threadIdx.x;  // (bi, j4)
  if (tid >= (size_t)BSZ * HDIM * 256) return;
  int    j4 = (int)(tid & 255);
  size_t bi = tid >> 8;                   // b*H + i
  int    i  = (int)(bi & (HDIM - 1));
  const float* er = eps + bi * HP1;
  int j = j4 * 4;
#pragma unroll
  for (int e = 0; e < 4; ++e) {
    int jj = j + e;
    float s = __expf(0.5f * logvar[(size_t)i * HP1 + jj]);
    Wm[bi * HDIM + jj] = (_Float16)(mu[(size_t)i * HP1 + jj] + s * er[jj]);
  }
}

__global__ void prep_wlast_kernel(const float* __restrict__ eps,
                                  const float* __restrict__ mu,
                                  const float* __restrict__ logvar,
                                  float* __restrict__ Wl) {
  int bi = blockIdx.x * blockDim.x + threadIdx.x;
  if (bi >= BSZ * HDIM) return;
  int i = bi & (HDIM - 1);
  float s = __expf(0.5f * logvar[(size_t)i * HP1 + HDIM]);
  Wl[bi] = mu[(size_t)i * HP1 + HDIM] + s * eps[(size_t)bi * HP1 + HDIM];
}

__global__ void cvt_f16_kernel(const float* __restrict__ src,
                               _Float16* __restrict__ dst, int n) {
  int i = blockIdx.x * blockDim.x + threadIdx.x;
  if (i < n) dst[i] = (_Float16)src[i];
}

// =====================================================================
// Encoder stage 1: t1[b,i] = relu( X0[b,:] . enc_w1[i,:] + b1[i] )  (K=32)
// =====================================================================
__global__ void enc1_kernel(const float* __restrict__ X0,
                            const float* __restrict__ w1,
                            const float* __restrict__ b1,
                            _Float16* __restrict__ t1h) {
  int idx = blockIdx.x * blockDim.x + threadIdx.x;   // b*H + i
  if (idx >= BSZ * HDIM) return;
  int b = idx >> 10, i = idx & (HDIM - 1);
  const float* x = X0 + b * OUTD;
  const float* w = w1 + (size_t)i * OUTD;
  float s = b1[i];
#pragma unroll
  for (int o = 0; o < OUTD; ++o) s = fmaf(x[o], w[o], s);
  t1h[idx] = (_Float16)fmaxf(s, 0.0f);
}

// =====================================================================
// GRU gates: gi computed inline (K=8), gh from GEMM (b_hh already added).
// Writes hc as fp16 (the "f" vector, bias column handled via Wlast).
// =====================================================================
__global__ void gru_gates_kernel(const float* __restrict__ At,
                                 const float* __restrict__ w_ih,
                                 const float* __restrict__ b_ih,
                                 const float* __restrict__ gh,
                                 const float* __restrict__ hprev,
                                 _Float16* __restrict__ hc_h) {
  int idx = blockIdx.x * blockDim.x + threadIdx.x;   // b*H + i
  if (idx >= BSZ * HDIM) return;
  int b = idx >> 10, i = idx & (HDIM - 1);
  const float* a = At + b * ACTD;
  float g[3];
#pragma unroll
  for (int q = 0; q < 3; ++q) {
    int row = i + q * HDIM;
    const float* w = w_ih + (size_t)row * ACTD;
    float s = b_ih[row];
#pragma unroll
    for (int j = 0; j < ACTD; ++j) s = fmaf(a[j], w[j], s);
    g[q] = s;
  }
  const float* ghr = gh + (size_t)b * 3 * HDIM;
  float r = 1.0f / (1.0f + __expf(-(g[0] + ghr[i])));
  float z = 1.0f / (1.0f + __expf(-(g[1] + ghr[i + HDIM])));
  float n = tanhf(g[2] + r * ghr[i + 2 * HDIM]);
  float hc = (1.0f - z) * n + z * hprev[idx];
  hc_h[idx] = (_Float16)hc;
}

// =====================================================================
// Bandwidth king: per-sample matvec  nh[b,i] = tanh( Wm[b,i,:].hc[b,:] + Wl[b,i] )
// One wave per row; lanes split K with 16B v8h loads; fp32 accumulate.
// 256 MiB read per timestep -> HBM-limited.
// =====================================================================
__global__ void bayes_matvec_kernel(const _Float16* __restrict__ Wm,
                                    const float* __restrict__ Wl,
                                    const _Float16* __restrict__ hc_h,
                                    float* __restrict__ h_out,
                                    _Float16* __restrict__ h_out_h) {
  __shared__ __align__(16) _Float16 fbuf[HDIM];
  int b    = blockIdx.x >> 7;        // 128 blocks per batch element
  int rblk = blockIdx.x & 127;
  if (threadIdx.x < 128)
    ((v8h*)fbuf)[threadIdx.x] = ((const v8h*)(hc_h + (size_t)b * HDIM))[threadIdx.x];
  __syncthreads();

  int wave = threadIdx.x >> 5, lane = threadIdx.x & 31;
  int i = rblk * 8 + wave;
  const _Float16* row = Wm + ((size_t)b * HDIM + i) * HDIM;
  float acc = 0.0f;
#pragma unroll
  for (int it = 0; it < 4; ++it) {
    int j = it * 256 + lane * 8;
    v8h wv = *(const v8h*)(row + j);
    v8h fv = *(const v8h*)(fbuf + j);
#pragma unroll
    for (int e = 0; e < 8; ++e) acc = fmaf((float)wv[e], (float)fv[e], acc);
  }
  acc = wave_reduce(acc);
  if (lane == 0) {
    float v = tanhf(acc + Wl[(size_t)b * HDIM + i]);
    h_out  [(size_t)b * HDIM + i] = v;
    h_out_h[(size_t)b * HDIM + i] = (_Float16)v;
  }
}

// =====================================================================
// Log-likelihood for one timestep (single block, deterministic accumulate)
// =====================================================================
__global__ void ll_kernel(const float* __restrict__ mean,
                          const float* __restrict__ lv,
                          const float* __restrict__ Xt,
                          float* __restrict__ acc) {
  __shared__ float sred[32];
  float s = 0.0f;
  for (int idx = threadIdx.x; idx < BSZ * OUTD; idx += 1024) {
    float m = mean[idx], l = lv[idx], x = Xt[idx];
    float d = x - m;
    s += -0.5f * d * d * __expf(-l) - 0.5f * l - 0.5f * LOG2PI_F;
  }
  s = wave_reduce(s);
  if ((threadIdx.x & 31) == 0) sred[threadIdx.x >> 5] = s;
  __syncthreads();
  if (threadIdx.x == 0) {
    float t = 0.0f;
#pragma unroll
    for (int w = 0; w < 32; ++w) t += sred[w];
    acc[0] += t;   // single-thread RMW, stream-serialized -> deterministic
  }
}

// ---------------- KL (two-stage deterministic reduction) ----------------
__global__ void kl_part_kernel(const float* __restrict__ qmu,
                               const float* __restrict__ qlv,
                               const float* __restrict__ pmu,
                               const float* __restrict__ plv,
                               float* __restrict__ part) {
  __shared__ float sred[8];
  float s = 0.0f;
  const int total = HDIM * HP1;
  for (int idx = blockIdx.x * blockDim.x + threadIdx.x; idx < total;
       idx += gridDim.x * blockDim.x) {
    float ql = qlv[idx], pl = plv[idx];
    float dm = qmu[idx] - pmu[idx];
    s += 0.5f * (pl - ql) + (__expf(ql) + dm * dm) / (2.0f * __expf(pl)) - 0.5f;
  }
  s = wave_reduce(s);
  if ((threadIdx.x & 31) == 0) sred[threadIdx.x >> 5] = s;
  __syncthreads();
  if (threadIdx.x == 0) {
    float t = 0.0f;
#pragma unroll
    for (int w = 0; w < 8; ++w) t += sred[w];
    part[blockIdx.x] = t;
  }
}

__global__ void kl_finish_kernel(const float* __restrict__ part, float* __restrict__ acc) {
  __shared__ float sred[8];
  float s = 0.0f;
  for (int i = threadIdx.x; i < 2048; i += 256) s += part[i];
  s = wave_reduce(s);
  if ((threadIdx.x & 31) == 0) sred[threadIdx.x >> 5] = s;
  __syncthreads();
  if (threadIdx.x == 0) {
    float t = 0.0f;
#pragma unroll
    for (int w = 0; w < 8; ++w) t += sred[w];
    acc[1] = t;
  }
}

__global__ void init_acc_kernel(float* acc) {
  if (threadIdx.x < 2) acc[threadIdx.x] = 0.0f;
}

__global__ void finalize_kernel(const float* __restrict__ acc, float* __restrict__ out) {
  float LLn = acc[0] / (float)BSZ;
  float KLn = acc[1] / ((float)TSTEPS * (float)BSZ);
  out[0] = LLn - KLn;  // FE
  out[1] = LLn;
  out[2] = KLn;
}

// =====================================================================
// Host orchestration
// =====================================================================
extern "C" void kernel_launch(void* const* d_in, const int* in_sizes, int n_in,
                              void* d_out, int out_size, void* d_ws, size_t ws_size,
                              hipStream_t stream) {
  (void)in_sizes; (void)n_in; (void)out_size; (void)ws_size;
  const float* X       = (const float*)d_in[0];
  const float* A       = (const float*)d_in[1];
  const float* W_eps   = (const float*)d_in[2];
  const float* W_mu    = (const float*)d_in[3];
  const float* W_lv    = (const float*)d_in[4];
  const float* pW_mu   = (const float*)d_in[5];
  const float* pW_lv   = (const float*)d_in[6];
  const float* enc_w1  = (const float*)d_in[7];
  const float* enc_b1  = (const float*)d_in[8];
  const float* enc_w2  = (const float*)d_in[9];
  const float* enc_b2  = (const float*)d_in[10];
  const float* gw_ih   = (const float*)d_in[11];
  const float* gw_hh   = (const float*)d_in[12];
  const float* gb_ih   = (const float*)d_in[13];
  const float* gb_hh   = (const float*)d_in[14];
  const float* dec_w   = (const float*)d_in[15];
  const float* dec_b   = (const float*)d_in[16];
  const float* em_w    = (const float*)d_in[17];
  const float* em_b    = (const float*)d_in[18];
  const float* ev_w    = (const float*)d_in[19];
  const float* ev_b    = (const float*)d_in[20];
  float* out = (float*)d_out;

  // ---- workspace carve (256B aligned) ----
  char* ws = (char*)d_ws;
  auto carve = [&](size_t bytes) -> void* {
    void* p = (void*)ws;
    ws += (bytes + 255) & ~(size_t)255;
    return p;
  };
  _Float16* Wm       = (_Float16*)carve((size_t)BSZ * HDIM * HDIM * 2); // 256 MiB
  float*    Wl       = (float*)   carve((size_t)BSZ * HDIM * 4);
  _Float16* w_hh_h   = (_Float16*)carve((size_t)3 * HDIM * HDIM * 2);
  _Float16* dec_w_h  = (_Float16*)carve((size_t)HDIM * HDIM * 2);
  _Float16* enc_w2_h = (_Float16*)carve((size_t)HDIM * HDIM * 2);
  _Float16* em_w_h   = (_Float16*)carve((size_t)OUTD * HDIM * 2);
  _Float16* ev_w_h   = (_Float16*)carve((size_t)OUTD * HDIM * 2);
  float*    h        = (float*)   carve((size_t)BSZ * HDIM * 4);
  _Float16* h_h      = (_Float16*)carve((size_t)BSZ * HDIM * 2);
  float*    gh       = (float*)   carve((size_t)BSZ * 3 * HDIM * 4);
  _Float16* hc_h     = (_Float16*)carve((size_t)BSZ * HDIM * 2);
  float*    decbuf   = (float*)   carve((size_t)BSZ * HDIM * 4);
  _Float16* dec_h    = (_Float16*)carve((size_t)BSZ * HDIM * 2);
  _Float16* t1_h     = (_Float16*)carve((size_t)BSZ * HDIM * 2);
  float*    meanb    = (float*)   carve((size_t)BSZ * OUTD * 4);
  float*    lvb      = (float*)   carve((size_t)BSZ * OUTD * 4);
  float*    klpart   = (float*)   carve(2048 * 4);
  float*    acc      = (float*)   carve(256);

  // ---- one-time prep ----
  init_acc_kernel<<<1, 32, 0, stream>>>(acc);

  auto cvt = [&](const float* s, _Float16* d, int n) {
    cvt_f16_kernel<<<(n + 255) / 256, 256, 0, stream>>>(s, d, n);
  };
  cvt(gw_hh,  w_hh_h,   3 * HDIM * HDIM);
  cvt(dec_w,  dec_w_h,  HDIM * HDIM);
  cvt(enc_w2, enc_w2_h, HDIM * HDIM);
  cvt(em_w,   em_w_h,   OUTD * HDIM);
  cvt(ev_w,   ev_w_h,   OUTD * HDIM);

  prep_w_kernel<<<(BSZ * HDIM * 256) / 256, 256, 0, stream>>>(W_eps, W_mu, W_lv, Wm);
  prep_wlast_kernel<<<(BSZ * HDIM) / 256, 256, 0, stream>>>(W_eps, W_mu, W_lv, Wl);

  kl_part_kernel<<<2048, 256, 0, stream>>>(W_mu, W_lv, pW_mu, pW_lv, klpart);
  kl_finish_kernel<<<1, 256, 0, stream>>>(klpart, acc);

  // ---- encoder: h0 = tanh(relu(X0 @ w1^T + b1) @ w2^T + b2) ----
  enc1_kernel<<<(BSZ * HDIM) / 256, 256, 0, stream>>>(X, enc_w1, enc_b1, t1_h);
  gemm_nt_wmma<<<dim3(HDIM / 64, BSZ / 16), 128, 0, stream>>>(
      t1_h, HDIM, enc_w2_h, HDIM, enc_b2, h, HDIM, h_h, HDIM, HDIM, HDIM, /*tanh*/2);

  // ---- sequential scan ----
  for (int t = 0; t < TSTEPS; ++t) {
    const float* At = A + (size_t)t * BSZ * ACTD;
    const float* Xt = X + (size_t)(t + 1) * BSZ * OUTD;

    // gh = h @ w_hh^T + b_hh   (128 x 3072, K=1024)  [WMMA]
    gemm_nt_wmma<<<dim3(3 * HDIM / 64, BSZ / 16), 128, 0, stream>>>(
        h_h, HDIM, w_hh_h, HDIM, gb_hh, gh, 3 * HDIM, (_Float16*)nullptr, 0,
        3 * HDIM, HDIM, 0);

    // GRU gates -> hc (fp16)
    gru_gates_kernel<<<(BSZ * HDIM) / 256, 256, 0, stream>>>(
        At, gw_ih, gb_ih, gh, h, hc_h);

    // nh = tanh(W[b] @ [hc;1])  -> updates h, h_h   [HBM-bound fp16 stream]
    bayes_matvec_kernel<<<BSZ * 128, 256, 0, stream>>>(Wm, Wl, hc_h, h, h_h);

    // dec = relu(nh @ dec_w^T + dec_b)   [WMMA]
    gemm_nt_wmma<<<dim3(HDIM / 64, BSZ / 16), 128, 0, stream>>>(
        h_h, HDIM, dec_w_h, HDIM, dec_b, decbuf, HDIM, dec_h, HDIM,
        HDIM, HDIM, /*relu*/1);

    // mean / logvar heads (128 x 32, K=1024)   [WMMA]
    gemm_nt_wmma<<<dim3(1, BSZ / 16), 128, 0, stream>>>(
        dec_h, HDIM, em_w_h, HDIM, em_b, meanb, OUTD, (_Float16*)nullptr, 0,
        OUTD, HDIM, 0);
    gemm_nt_wmma<<<dim3(1, BSZ / 16), 128, 0, stream>>>(
        dec_h, HDIM, ev_w_h, HDIM, ev_b, lvb, OUTD, (_Float16*)nullptr, 0,
        OUTD, HDIM, 0);

    // accumulate log-likelihood (deterministic)
    ll_kernel<<<1, 1024, 0, stream>>>(meanb, lvb, Xt, acc);
  }

  finalize_kernel<<<1, 1, 0, stream>>>(acc, out);
}
